// SMTRLocalDecoder_23347442221449
// MI455X (gfx1250) — matrix operand order repaired and checked by reading, not compiled, not run on hardware
//
#include <hip/hip_runtime.h>
#include <math.h>

// ---------------------------------------------------------------------------
// SMTRLocalDecoder for MI455X (gfx1250, wave32, WMMA).
// All GEMM-like contractions run through v_wmma_f32_16x16x32_bf16 with fp32
// accumulation; fp32 operands are converted to bf16 inside the fragment
// loaders with wide (b64/b128) loads. Register blocking (1 A-frag feeding 4+
// WMMAs) keeps the XDL pipe fed. Softmax / LayerNorm / GELU / sigmoid in fp32.
// ---------------------------------------------------------------------------

typedef __attribute__((ext_vector_type(16))) __bf16 v16bf;
typedef __attribute__((ext_vector_type(8)))  float  v8f;

#define DEVINL __device__ __forceinline__

// Problem constants
#define BS_  48
#define H_   8
#define W_   256
#define S_   64
#define C_   384
#define HD_  192
#define KVLD_ 768
#define RS192 0.07216878364870323f   /* 192^-0.5 */

// ---- WMMA helpers ---------------------------------------------------------

DEVINL v8f wmma_bf16(v16bf a, v16bf b, v8f c) {
  return __builtin_amdgcn_wmma_f32_16x16x32_bf16(false, a, false, b, (short)0, c,
                                                 false, false);
}

DEVINL v8f vzero8() {
  v8f z = {0.f, 0.f, 0.f, 0.f, 0.f, 0.f, 0.f, 0.f};
  return z;
}

// A-matrix 16x32 bf16 layout: lanes 0-15 hold K 0..15,
// lanes 16-31 hold K 16..31 (matches B); element pairs pack into dwords.
// Pairwise-contiguous -> 8 x b64 loads.
DEVINL v16bf loadA_f32(const float* __restrict__ p, int ld, int row0, int k0,
                       int lane) {
  int half = lane >> 4, idx = lane & 15;
  const float* rp = p + (long)(row0 + idx) * ld + (k0 + 16 * half);
  v16bf f;
#pragma unroll
  for (int g = 0; g < 8; ++g) {
    int k = 2 * g;
    float2 v = *(const float2*)(rp + k);
    f[2 * g]     = (__bf16)v.x;
    f[2 * g + 1] = (__bf16)v.y;
  }
  return f;
}

// A fragment from LDS bf16 probs [16, ld] (row = lane&15), 8 x b32 loads.
DEVINL v16bf loadA_bf16(const unsigned short* __restrict__ p, int ld, int k0,
                        int lane) {
  int half = lane >> 4, idx = lane & 15;
  const unsigned int* rp =
      (const unsigned int*)(p + idx * ld + (k0 + 16 * half));
  v16bf f;
#pragma unroll
  for (int g = 0; g < 8; ++g) {
    int kd = g;  // dword index = kmap/2
    unsigned int w = rp[kd];
    f[2 * g]     = __builtin_bit_cast(__bf16, (unsigned short)(w & 0xffffu));
    f[2 * g + 1] = __builtin_bit_cast(__bf16, (unsigned short)(w >> 16));
  }
  return f;
}

// B-matrix 32x16 bf16 layout (contiguous k order used here, same as A): lanes 0-15 -> K
// 0..15, lanes 16-31 -> K 16..31, lane = N. Memory row-major [n, ld] over k
// (weights stored (out,in); K-matrix rows). 16 contiguous K -> 4 x b128.
DEVINL v16bf loadB_f32(const float* __restrict__ p, int ld, int n0, int k0,
                       int lane) {
  int half = lane >> 4, idx = lane & 15;
  const float* rp = p + (long)(n0 + idx) * ld + (k0 + 16 * half);
  v16bf f;
#pragma unroll
  for (int q4 = 0; q4 < 4; ++q4) {
    float4 v = *(const float4*)(rp + 4 * q4);
    f[4 * q4 + 0] = (__bf16)v.x;
    f[4 * q4 + 1] = (__bf16)v.y;
    f[4 * q4 + 2] = (__bf16)v.z;
    f[4 * q4 + 3] = (__bf16)v.w;
  }
  return f;
}

// B fragment where memory is row-major [k, ld] over n (V matrix): per-lane
// column walk, scalar loads (coalesced across lanes).
DEVINL v16bf loadB_f32_T(const float* __restrict__ p, int ld, int k0, int n0,
                         int lane) {
  int half = lane >> 4, idx = lane & 15;
  const float* cp = p + (long)(k0 + 16 * half) * ld + (n0 + idx);
  v16bf f;
#pragma unroll
  for (int e = 0; e < 16; ++e) f[e] = (__bf16)cp[(long)e * ld];
  return f;
}

// ---- Generic GEMM: out[M,N] = act(A[M,K] @ W[N,K]^T + bias) * scale --------
// One wave computes a 16x64 tile (4 N-subtiles) so each A fragment feeds 4
// WMMAs.

__global__ void gemm_bias_kernel(const float* __restrict__ A,
                                 const float* __restrict__ W,
                                 const float* __restrict__ bias,
                                 float* __restrict__ out, int M, int N, int K,
                                 float scale, int act) {
  int wave = blockIdx.x * (blockDim.x >> 5) + (threadIdx.x >> 5);
  int lane = threadIdx.x & 31;
  int tilesN4 = N >> 6;  // N/64
  int tot = (M >> 4) * tilesN4;
  if (wave >= tot) return;  // wave-uniform: EXEC all-ones at every WMMA
  int tm = wave / tilesN4, tn4 = wave - tm * tilesN4;
  v8f acc[4] = {vzero8(), vzero8(), vzero8(), vzero8()};
  for (int k0 = 0; k0 < K; k0 += 32) {
    v16bf fa = loadA_f32(A, K, tm * 16, k0, lane);
#pragma unroll
    for (int j = 0; j < 4; ++j) {
      v16bf fb = loadB_f32(W, K, (tn4 * 4 + j) * 16, k0, lane);
      acc[j] = wmma_bf16(fa, fb, acc[j]);
    }
  }
  int half = lane >> 4, idx = lane & 15;
#pragma unroll
  for (int j = 0; j < 4; ++j) {
#pragma unroll
    for (int r = 0; r < 8; ++r) {
      int row = tm * 16 + r + 8 * half;  // C/D layout (ISA 7.12.2)
      int col = (tn4 * 4 + j) * 16 + idx;
      float v = acc[j][r] + bias[col];
      if (act == 1) v = tanhf(v);
      out[(long)row * N + col] = v * scale;
    }
  }
}

// ---- RCTC stand-in: y[b,w,c] = mean_h GELU(kv[b,h,w,:] @ rctc_w^T + b) -----
// Loop order puts K outermost so one weight B-fragment feeds 8 WMMAs (one per
// h) into 8 independent accumulators.

__global__ void rctc_mean_kernel(const float* __restrict__ kv,
                                 const float* __restrict__ Wt,
                                 const float* __restrict__ bias,
                                 float* __restrict__ y) {
  const int TW = W_ / 16, TC = C_ / 16;  // 16 x 24
  int wave = blockIdx.x * (blockDim.x >> 5) + (threadIdx.x >> 5);
  int lane = threadIdx.x & 31;
  int tot = BS_ * TW * TC;
  if (wave >= tot) return;
  int b = wave / (TW * TC);
  int rem = wave - b * (TW * TC);
  int wt = rem / TC, ct = rem - wt * TC;
  int half = lane >> 4, idx = lane & 15;
  v8f acc[H_];
#pragma unroll
  for (int h = 0; h < H_; ++h) acc[h] = vzero8();
  for (int k0 = 0; k0 < C_; k0 += 32) {
    v16bf fb = loadB_f32(Wt, C_, ct * 16, k0, lane);
#pragma unroll
    for (int h = 0; h < H_; ++h) {
      v16bf fa = loadA_f32(kv, C_, (b * H_ + h) * W_ + wt * 16, k0, lane);
      acc[h] = wmma_bf16(fa, fb, acc[h]);
    }
  }
#pragma unroll
  for (int r = 0; r < 8; ++r) {
    float bb = bias[ct * 16 + idx];
    float s = 0.f;
#pragma unroll
    for (int h = 0; h < H_; ++h) {
      float v = acc[h][r] + bb;
      v = 0.5f * v * (1.0f + erff(v * 0.70710678118654752f));  // exact GELU
      s += v;
    }
    int wrow = wt * 16 + r + 8 * half;
    y[((long)b * W_ + wrow) * C_ + ct * 16 + idx] = s * 0.125f;
  }
}

// ---- LayerNorm over C=384, one wave per row --------------------------------

__global__ void ln_kernel(const float* __restrict__ x,
                          const float* __restrict__ s,
                          const float* __restrict__ b, float* __restrict__ o,
                          int rows) {
  int row = blockIdx.x * (blockDim.x >> 5) + (threadIdx.x >> 5);
  if (row >= rows) return;
  int lane = threadIdx.x & 31;
  const float* xr = x + (long)row * C_;
  float vals[12];
  float sum = 0.f;
#pragma unroll
  for (int i = 0; i < 12; ++i) {
    vals[i] = xr[lane + i * 32];
    sum += vals[i];
  }
#pragma unroll
  for (int m = 16; m; m >>= 1) sum += __shfl_xor(sum, m, 32);
  float mu = sum * (1.0f / 384.0f);
  float var = 0.f;
#pragma unroll
  for (int i = 0; i < 12; ++i) {
    float d = vals[i] - mu;
    var += d * d;
  }
#pragma unroll
  for (int m = 16; m; m >>= 1) var += __shfl_xor(var, m, 32);
  var *= (1.0f / 384.0f);
  float rstd = rsqrtf(var + 1e-6f);
  float* orow = o + (long)row * C_;
#pragma unroll
  for (int i = 0; i < 12; ++i) {
    int c = lane + i * 32;
    orow[c] = (vals[i] - mu) * rstd * s[c] + b[c];
  }
}

// ---- p_t head: sigmoid(t . vp_w + vp_b) * W, one wave per row --------------

__global__ void pt_kernel(const float* __restrict__ t,
                          const float* __restrict__ vpw,
                          const float* __restrict__ vpb,
                          float* __restrict__ pt, int rows) {
  int row = blockIdx.x * (blockDim.x >> 5) + (threadIdx.x >> 5);
  if (row >= rows) return;
  int lane = threadIdx.x & 31;
  const float* tr = t + (long)row * C_;
  float sum = 0.f;
#pragma unroll
  for (int i = 0; i < 12; ++i) sum += tr[lane + i * 32] * vpw[lane + i * 32];
#pragma unroll
  for (int m = 16; m; m >>= 1) sum += __shfl_xor(sum, m, 32);
  if (lane == 0) {
    float x = sum + vpb[0];
    pt[row] = (float)W_ / (1.0f + __expf(-x));
  }
}

// ---- Generic attention (heads=2, hd=192): one wave per (b, head, qtile) ----
// Q:  [B*QN, 384], head channels at head*192
// KV: [B*KN, 768], K at head*192, V at 384 + head*192
// PT: optional [B*QN] Gaussian-window centers (final attention only)
// X:  [B*QN, 384] output (channel = head*192 + d)

__global__ void attn_kernel(const float* __restrict__ Q,
                            const float* __restrict__ KV,
                            const float* __restrict__ PT,
                            float* __restrict__ X, int B, int QN, int KN) {
  int qtiles = QN >> 4;
  int bid = blockIdx.x;
  int qt = bid % qtiles;
  bid /= qtiles;
  int head = bid & 1;
  int b = bid >> 1;
  int lane = threadIdx.x;  // blockDim.x == 32
  int half = lane >> 4, idx = lane & 15;

  extern __shared__ char smem[];
  float* sc = (float*)smem;                                    // [16, KN] fp32
  unsigned short* pb = (unsigned short*)(smem + (size_t)16 * KN * 4);  // bf16

  const float* qp = Q + (long)b * QN * C_ + head * HD_;
  const float* kp = KV + (long)b * KN * KVLD_ + head * HD_;
  const float* vp = KV + (long)b * KN * KVLD_ + C_ + head * HD_;

  // ---- scores: S[16, KN] = (Q tile) @ K^T, optional Gaussian window -------
  // 4 K-tiles per pass so each Q fragment feeds 4 WMMAs.
  for (int tb = 0; tb < (KN >> 4); tb += 4) {
    v8f acc[4] = {vzero8(), vzero8(), vzero8(), vzero8()};
    for (int k0 = 0; k0 < HD_; k0 += 32) {
      v16bf fa = loadA_f32(qp, C_, qt * 16, k0, lane);
#pragma unroll
      for (int j = 0; j < 4; ++j) {
        v16bf fb = loadB_f32(kp, KVLD_, (tb + j) * 16, k0, lane);
        acc[j] = wmma_bf16(fa, fb, acc[j]);
      }
    }
#pragma unroll
    for (int j = 0; j < 4; ++j) {
#pragma unroll
      for (int r = 0; r < 8; ++r) {
        int qrow = r + 8 * half;  // local row in tile
        int kn = (tb + j) * 16 + idx;
        float v = acc[j][r];
        if (PT) {
          float d = (float)kn - PT[b * QN + qt * 16 + qrow];
          v *= __expf(d * d * (-1.0f / 18.0f));  // 2*sigma^2 = 18
        }
        sc[qrow * KN + kn] = v;
      }
    }
  }
  __syncthreads();

  // ---- fp32 softmax over KN, 2 lanes per row; probs packed to bf16 --------
  {
    int row = lane >> 1, part = lane & 1;
    float m = -1e30f;
    for (int kn = part; kn < KN; kn += 2) m = fmaxf(m, sc[row * KN + kn]);
    m = fmaxf(m, __shfl_xor(m, 1, 32));
    float ssum = 0.f;
    for (int kn = part; kn < KN; kn += 2) ssum += __expf(sc[row * KN + kn] - m);
    ssum += __shfl_xor(ssum, 1, 32);
    float inv = 1.0f / ssum;
    for (int kn = part; kn < KN; kn += 2) {
      __bf16 h = (__bf16)(__expf(sc[row * KN + kn] - m) * inv);
      pb[row * KN + kn] = __builtin_bit_cast(unsigned short, h);
    }
  }
  __syncthreads();

  // ---- X[16, 192] = probs @ V, 4 d-tiles per pass --------------------------
  for (int nb = 0; nb < (HD_ >> 4); nb += 4) {
    v8f acc[4] = {vzero8(), vzero8(), vzero8(), vzero8()};
    for (int k0 = 0; k0 < KN; k0 += 32) {
      v16bf fa = loadA_bf16(pb, KN, k0, lane);
#pragma unroll
      for (int j = 0; j < 4; ++j) {
        v16bf fb = loadB_f32_T(vp, KVLD_, k0, (nb + j) * 16, lane);
        acc[j] = wmma_bf16(fa, fb, acc[j]);
      }
    }
#pragma unroll
    for (int j = 0; j < 4; ++j) {
#pragma unroll
      for (int r = 0; r < 8; ++r) {
        int qrow = qt * 16 + r + 8 * half;
        X[((long)b * QN + qrow) * C_ + head * HD_ + (nb + j) * 16 + idx] =
            acc[j][r];
      }
    }
  }
}

// ---------------------------------------------------------------------------

extern "C" void kernel_launch(void* const* d_in, const int* in_sizes, int n_in,
                              void* d_out, int out_size, void* d_ws,
                              size_t ws_size, hipStream_t stream) {
  (void)in_sizes; (void)n_in; (void)out_size; (void)ws_size;
  const float* q      = (const float*)d_in[0];
  const float* kv     = (const float*)d_in[1];
  // d_in[2]=H, d_in[3]=W (scalars; hardcoded)
  const float* rctc_w = (const float*)d_in[4];
  const float* rctc_b = (const float*)d_in[5];
  const float* q_w    = (const float*)d_in[6];
  const float* q_b    = (const float*)d_in[7];
  const float* kv_w   = (const float*)d_in[8];
  const float* kv_b   = (const float*)d_in[9];
  const float* proj_w = (const float*)d_in[10];
  const float* proj_b = (const float*)d_in[11];
  const float* qpos_w = (const float*)d_in[12];
  const float* qpos_b = (const float*)d_in[13];
  const float* Wp_w   = (const float*)d_in[14];
  const float* Wp_b   = (const float*)d_in[15];
  const float* vp_w   = (const float*)d_in[16];
  const float* vp_b   = (const float*)d_in[17];
  const float* nq1_s  = (const float*)d_in[18];
  const float* nq1_b  = (const float*)d_in[19];
  const float* nk1_s  = (const float*)d_in[20];
  const float* nk1_b  = (const float*)d_in[21];
  const float* nq2_s  = (const float*)d_in[22];
  const float* nq2_b  = (const float*)d_in[23];
  const float* nk2_s  = (const float*)d_in[24];
  const float* nk2_b  = (const float*)d_in[25];
  const float* ca1_qw  = (const float*)d_in[26];
  const float* ca1_qb  = (const float*)d_in[27];
  const float* ca1_kvw = (const float*)d_in[28];
  const float* ca1_kvb = (const float*)d_in[29];
  const float* ca1_pw  = (const float*)d_in[30];
  const float* ca1_pb  = (const float*)d_in[31];
  const float* ca2_qw  = (const float*)d_in[32];
  const float* ca2_qb  = (const float*)d_in[33];
  const float* ca2_kvw = (const float*)d_in[34];
  const float* ca2_kvb = (const float*)d_in[35];
  const float* ca2_pw  = (const float*)d_in[36];
  const float* ca2_pb  = (const float*)d_in[37];

  float* ws = (float*)d_ws;
  size_t off = 0;
  auto alloc = [&](size_t n) { float* p = ws + off; off += n; return p; };
  const size_t SC = (size_t)BS_ * S_ * C_;      // 48*64*384
  float* y    = alloc((size_t)BS_ * W_ * C_);
  float* kvp  = alloc((size_t)BS_ * W_ * KVLD_);
  float* qh   = alloc(SC);
  float* p0   = alloc(SC);
  float* lnp0 = alloc(SC);
  float* lnq  = alloc(SC);
  float* q1   = alloc(SC);
  float* kv1  = alloc((size_t)BS_ * S_ * KVLD_);
  float* x1   = alloc(SC);
  float* p1   = alloc(SC);
  float* lnp1 = alloc(SC);
  float* lny  = alloc((size_t)BS_ * W_ * C_);
  float* q2   = alloc(SC);
  float* kv2  = alloc((size_t)BS_ * W_ * KVLD_);
  float* x2   = alloc(SC);
  float* p2   = alloc(SC);
  float* tt   = alloc(SC);
  float* xf   = alloc(SC);
  float* ptv  = alloc((size_t)BS_ * S_);
  float* outp = (float*)d_out;

  const dim3 blk(128);  // 4 waves per workgroup
  auto gemmGrid = [](int M, int N) {
    int t = (M / 16) * (N / 64);
    return dim3((unsigned)((t + 3) / 4));
  };
  auto rowGrid = [](int rows) { return dim3((unsigned)((rows + 3) / 4)); };

  // 1. y = mean_h GELU(kv @ rctc_w^T + rctc_b)
  rctc_mean_kernel<<<dim3((BS_ * (W_ / 16) * (C_ / 16) + 3) / 4), blk, 0,
                     stream>>>(kv, rctc_w, rctc_b, y);
  // 2. kvp = y @ kv_w^T + kv_b
  gemm_bias_kernel<<<gemmGrid(BS_ * W_, KVLD_), blk, 0, stream>>>(
      y, kv_w, kv_b, kvp, BS_ * W_, KVLD_, C_, 1.0f, 0);
  // 3. qh = (q @ q_w^T + q_b) * hd^-0.5
  gemm_bias_kernel<<<gemmGrid(BS_ * S_, C_), blk, 0, stream>>>(
      q, q_w, q_b, qh, BS_ * S_, C_, C_, RS192, 0);
  // 4. p0 = q @ qpos_w^T + qpos_b
  gemm_bias_kernel<<<gemmGrid(BS_ * S_, C_), blk, 0, stream>>>(
      q, qpos_w, qpos_b, p0, BS_ * S_, C_, C_, 1.0f, 0);
  // 5. LN inputs for cross-attn 1
  ln_kernel<<<rowGrid(BS_ * S_), blk, 0, stream>>>(p0, nq1_s, nq1_b, lnp0,
                                                   BS_ * S_);
  ln_kernel<<<rowGrid(BS_ * S_), blk, 0, stream>>>(q, nk1_s, nk1_b, lnq,
                                                   BS_ * S_);
  // 6-7. cross-attn 1 projections
  gemm_bias_kernel<<<gemmGrid(BS_ * S_, C_), blk, 0, stream>>>(
      lnp0, ca1_qw, ca1_qb, q1, BS_ * S_, C_, C_, RS192, 0);
  gemm_bias_kernel<<<gemmGrid(BS_ * S_, KVLD_), blk, 0, stream>>>(
      lnq, ca1_kvw, ca1_kvb, kv1, BS_ * S_, KVLD_, C_, 1.0f, 0);
  // 8. cross-attn 1 core (KN = 64)
  attn_kernel<<<dim3(BS_ * 2 * (S_ / 16)), dim3(32), 16 * S_ * 6, stream>>>(
      q1, kv1, nullptr, x1, BS_, S_, S_);
  // 9. p1 = x1 @ ca1_pw^T + ca1_pb
  gemm_bias_kernel<<<gemmGrid(BS_ * S_, C_), blk, 0, stream>>>(
      x1, ca1_pw, ca1_pb, p1, BS_ * S_, C_, C_, 1.0f, 0);
  // 10. LN inputs for cross-attn 2
  ln_kernel<<<rowGrid(BS_ * S_), blk, 0, stream>>>(p1, nq2_s, nq2_b, lnp1,
                                                   BS_ * S_);
  ln_kernel<<<rowGrid(BS_ * W_), blk, 0, stream>>>(y, nk2_s, nk2_b, lny,
                                                   BS_ * W_);
  // 11-12. cross-attn 2 projections
  gemm_bias_kernel<<<gemmGrid(BS_ * S_, C_), blk, 0, stream>>>(
      lnp1, ca2_qw, ca2_qb, q2, BS_ * S_, C_, C_, RS192, 0);
  gemm_bias_kernel<<<gemmGrid(BS_ * W_, KVLD_), blk, 0, stream>>>(
      lny, ca2_kvw, ca2_kvb, kv2, BS_ * W_, KVLD_, C_, 1.0f, 0);
  // 13. cross-attn 2 core (KN = 256)
  attn_kernel<<<dim3(BS_ * 2 * (S_ / 16)), dim3(32), 16 * W_ * 6, stream>>>(
      q2, kv2, nullptr, x2, BS_, S_, W_);
  // 14. p2 = x2 @ ca2_pw^T + ca2_pb
  gemm_bias_kernel<<<gemmGrid(BS_ * S_, C_), blk, 0, stream>>>(
      x2, ca2_pw, ca2_pb, p2, BS_ * S_, C_, C_, 1.0f, 0);
  // 15. tt = tanh(p2 @ Wp_w^T + Wp_b)
  gemm_bias_kernel<<<gemmGrid(BS_ * S_, C_), blk, 0, stream>>>(
      p2, Wp_w, Wp_b, tt, BS_ * S_, C_, C_, 1.0f, 1);
  // 16. p_t = sigmoid(tt . vp_w + vp_b) * W
  pt_kernel<<<rowGrid(BS_ * S_), blk, 0, stream>>>(tt, vp_w, vp_b, ptv,
                                                   BS_ * S_);
  // 17. Gaussian-windowed local attention (KN = 256)
  attn_kernel<<<dim3(BS_ * 2 * (S_ / 16)), dim3(32), 16 * W_ * 6, stream>>>(
      qh, kvp, ptv, xf, BS_, S_, W_);
  // 18. out = xf @ proj_w^T + proj_b
  gemm_bias_kernel<<<gemmGrid(BS_ * S_, C_), blk, 0, stream>>>(
      xf, proj_w, proj_b, outp, BS_ * S_, C_, C_, 1.0f, 0);
}
